// MultiLayerGNN_10385230921968
// MI455X (gfx1250) — matrix operand order repaired and checked by reading, not compiled, not run on hardware
//
#include <hip/hip_runtime.h>
#include <hip/hip_bf16.h>
#include <math.h>

// ---------------- problem constants ----------------
constexpr int Bc   = 8;
constexpr int Tc   = 32;
constexpr int Nc   = 128;
constexpr int Ec   = 2048;
constexpr int Fc   = 512;                 // IN_DIM == HID == OUT
constexpr int Gc   = Bc * Tc;             // 256 graphs
constexpr int EPG  = Ec + Nc;             // 2176 edges incl. self loops
constexpr float SLOPE = 0.2f;

// ---------------- vector types ----------------
typedef float  v8f    __attribute__((ext_vector_type(8)));
typedef __bf16 v16bf  __attribute__((ext_vector_type(16)));
typedef __bf16 bf16x8 __attribute__((ext_vector_type(8)));

__device__ __forceinline__ float gelu_exact(float x) {
  return 0.5f * x * (1.0f + erff(x * 0.70710678118654752f));
}

// ---------------- conversion kernels ----------------
__global__ void cvt_f32_to_bf16(const float* __restrict__ in, __bf16* __restrict__ out, size_t n) {
  size_t i = (size_t)blockIdx.x * blockDim.x + threadIdx.x;
  size_t stride = (size_t)gridDim.x * blockDim.x;
  for (; i < n; i += stride) out[i] = (__bf16)in[i];
}

// Wt[k*F + n] = W[n*F + k]  (so WMMA B fragment rows are contiguous)
__global__ void cvt_transpose_bf16(const float* __restrict__ W, __bf16* __restrict__ Wt) {
  size_t i = (size_t)blockIdx.x * blockDim.x + threadIdx.x;
  size_t n = (size_t)Fc * Fc;
  size_t stride = (size_t)gridDim.x * blockDim.x;
  for (; i < n; i += stride) {
    int k = (int)(i / Fc), c = (int)(i % Fc);
    Wt[i] = (__bf16)W[(size_t)c * Fc + k];
  }
}

// ---------------- WMMA GEMM: C[M x F] = A_bf[M x F] * Bt_bf (Bt[k][n]) ----------------
// Block: 8 waves, tile 128 rows x 64 cols. The 32(K)x64(N) B tile is shared by all
// waves and staged through LDS with double-buffered global_load_async_to_lds_b128
// (ASYNCcnt). The shared-array pointer is passed INTO the asm so the compiler knows
// the async op writes it (otherwise AA hoists the ds_loads as undef).
constexpr int KSTEP = 32;
constexpr int BROW  = 72;                  // padded LDS row stride (bf16 elems) -> 144B (16B-aligned rows)
constexpr int BUFE  = KSTEP * BROW;        // elems per buffer

__global__ void gemm_bf16_wmma(const __bf16* __restrict__ A, const __bf16* __restrict__ Bt,
                               float* __restrict__ C, int M) {
  __shared__ __bf16 shB[2 * BUFE];

  const int tid  = threadIdx.x;
  const int wave = tid >> 5;
  const int lane = tid & 31;
  const int half = lane >> 4;
  const int l15  = lane & 15;

  const int tiles_n = Fc / 64;             // 8
  const int bm = blockIdx.x / tiles_n;
  const int bn = blockIdx.x % tiles_n;
  const int row0 = bm * 128 + wave * 16;
  const int col0 = bn * 64;

  // cooperative staging: 256 threads x 16B = 32x64 bf16 tile
  const int srow = tid >> 3;               // 0..31  (k within tile)
  const int sseg = tid & 7;                // 0..7   (8 bf16 = 16B each)

  auto stage = [&](int buf, int k0) {
    const __bf16* g = Bt + (size_t)(k0 + srow) * Fc + col0 + sseg * 8;
    __bf16* l = &shB[buf * BUFE + srow * BROW + sseg * 8];
    // addr[31:0] of a generic LDS pointer is the wave-relative LDS byte address.
    unsigned lds = (unsigned)(uintptr_t)l;
    asm volatile("global_load_async_to_lds_b128 %0, %1, off"
                 :: "v"(lds), "v"(g), "v"(l) : "memory");
  };

  stage(0, 0);

  v8f acc0 = {}, acc1 = {}, acc2 = {}, acc3 = {};
  const __bf16* arowbase = A + (size_t)(row0 + l15) * Fc + half * 8;

  for (int i = 0; i < Fc / KSTEP; ++i) {
    asm volatile("s_wait_asynccnt 0x0" ::: "memory");
    __syncthreads();                       // current buffer fully in LDS, prev reads done
    if (i + 1 < Fc / KSTEP) stage((i + 1) & 1, (i + 1) * KSTEP);

    // ----- A fragment (ISA layout: e<8 -> k=half*8+e ; e>=8 -> k=16+half*8+(e-8))
    const __bf16* arow = arowbase + i * KSTEP;
    bf16x8 alo = *(const bf16x8*)(arow);
    bf16x8 ahi = *(const bf16x8*)(arow + 16);
    if (i + 1 < Fc / KSTEP) __builtin_prefetch((const void*)(arow + KSTEP), 0, 3);
    v16bf a;
#pragma unroll
    for (int q = 0; q < 8; ++q) { a[q] = alo[q]; a[8 + q] = ahi[q]; }

    // ----- all 4 B fragments from LDS first (clause the ds_loads), then 4 WMMAs
    const __bf16* bbase = &shB[(i & 1) * BUFE + l15];
    v16bf bf0, bf1, bf2, bf3;
#pragma unroll
    for (int nt = 0; nt < 4; ++nt) {
      v16bf& bfr = (nt == 0) ? bf0 : (nt == 1) ? bf1 : (nt == 2) ? bf2 : bf3;
#pragma unroll
      for (int q = 0; q < 8; ++q) { bfr[q] = bbase[(half * 8 + q) * BROW + nt * 16]; bfr[8 + q] = bbase[(16 + half * 8 + q) * BROW + nt * 16]; }
    }
    acc0 = __builtin_amdgcn_wmma_f32_16x16x32_bf16(false, a, false, bf0, (short)0, acc0, false, false);
    acc1 = __builtin_amdgcn_wmma_f32_16x16x32_bf16(false, a, false, bf1, (short)0, acc1, false, false);
    acc2 = __builtin_amdgcn_wmma_f32_16x16x32_bf16(false, a, false, bf2, (short)0, acc2, false, false);
    acc3 = __builtin_amdgcn_wmma_f32_16x16x32_bf16(false, a, false, bf3, (short)0, acc3, false, false);
  }

  // ----- store: row = row0 + r + 8*half, col = col0 + nt*16 + (lane&15)
#pragma unroll
  for (int nt = 0; nt < 4; ++nt) {
    const v8f& acc = (nt == 0) ? acc0 : (nt == 1) ? acc1 : (nt == 2) ? acc2 : acc3;
    float* crow = C + (size_t)(row0 + 8 * half) * Fc + col0 + nt * 16 + l15;
#pragma unroll
    for (int r = 0; r < 8; ++r) crow[(size_t)r * Fc] = acc[r];
  }
  (void)M;
}

// ---------------- per-node attention scores: ssrc = h.a_src, sdst = h.a_dst ----------------
__global__ void scores_kernel(const float* __restrict__ H,
                              const float* __restrict__ a_src, const float* __restrict__ a_dst,
                              float* __restrict__ ssrc, float* __restrict__ sdst) {
  const int lane = threadIdx.x & 31;
  const int node = blockIdx.x * (blockDim.x >> 5) + (threadIdx.x >> 5);   // one wave per node
  const float* h = H + (size_t)node * Fc;
  float s0 = 0.f, s1 = 0.f;
#pragma unroll
  for (int j = 0; j < Fc / 32; ++j) {
    float v = h[lane + 32 * j];
    s0 += v * a_src[lane + 32 * j];
    s1 += v * a_dst[lane + 32 * j];
  }
#pragma unroll
  for (int o = 16; o > 0; o >>= 1) {
    s0 += __shfl_xor(s0, o, 32);
    s1 += __shfl_xor(s1, o, 32);
  }
  if (lane == 0) { ssrc[node] = s0; sdst[node] = s1; }
}

// ---------------- CSR construction (shared by both layers) ----------------
__device__ __forceinline__ void edge_sd(const int* __restrict__ ei, int g, int e, int& s, int& d) {
  if (e < Ec) {
    s = ei[((size_t)g * 2 + 0) * Ec + e];
    d = ei[((size_t)g * 2 + 1) * Ec + e];
  } else {
    s = d = e - Ec;                                   // self loop
  }
}

__global__ void deg_kernel(const int* __restrict__ ei, int* __restrict__ deg) {
  size_t tid = (size_t)blockIdx.x * blockDim.x + threadIdx.x;
  if (tid >= (size_t)Gc * EPG) return;
  int g = (int)(tid / EPG), e = (int)(tid % EPG), s, d;
  edge_sd(ei, g, e, s, d);
  atomicAdd(&deg[g * Nc + d], 1);
}

__global__ void scan_kernel(const int* __restrict__ deg, int* __restrict__ off) {
  __shared__ int sm[Nc];
  int g = blockIdx.x, t = threadIdx.x;
  int d = deg[g * Nc + t];
  sm[t] = d; __syncthreads();
#pragma unroll
  for (int o = 1; o < Nc; o <<= 1) {
    int v = (t >= o) ? sm[t - o] : 0;
    __syncthreads();
    sm[t] += v;
    __syncthreads();
  }
  off[g * Nc + t] = sm[t] - d;                        // exclusive
}

__global__ void fill_kernel(const int* __restrict__ ei, const int* __restrict__ off,
                            int* __restrict__ cur, int* __restrict__ csr) {
  size_t tid = (size_t)blockIdx.x * blockDim.x + threadIdx.x;
  if (tid >= (size_t)Gc * EPG) return;
  int g = (int)(tid / EPG), e = (int)(tid % EPG), s, d;
  edge_sd(ei, g, e, s, d);
  int pos = atomicAdd(&cur[g * Nc + d], 1);
  csr[(size_t)g * EPG + off[g * Nc + d] + pos] = s;
}

// ---------------- fused GAT softmax + aggregation: one wave per dst node ----------------
// layer==0: out = gelu(agg + b)        -> bf16 Ybf only (feeds next GEMM)
// layer==1: out = (agg + b) * timemask -> f32 Yf32 and bf16 Ybf
__global__ void gat_agg_kernel(const float* __restrict__ H,
                               const float* __restrict__ ssrc, const float* __restrict__ sdst,
                               const int* __restrict__ off, const int* __restrict__ deg,
                               const int* __restrict__ csr,
                               const float* __restrict__ bias, const int* __restrict__ valid,
                               float* __restrict__ Yf32, __bf16* __restrict__ Ybf, int layer) {
  const int lane = threadIdx.x & 31;
  const int w = blockIdx.x * (blockDim.x >> 5) + (threadIdx.x >> 5);      // global node index
  const int g = w / Nc;
  const int d = deg[w];
  const int* edges = csr + (size_t)g * EPG + off[w];
  const float sd = sdst[w];

  // pass 1: segment max
  float mx = -3.4e38f;
  for (int j = lane; j < d; j += 32) {
    float e = ssrc[g * Nc + edges[j]] + sd;
    e = (e > 0.f) ? e : SLOPE * e;
    mx = fmaxf(mx, e);
  }
#pragma unroll
  for (int o = 16; o > 0; o >>= 1) mx = fmaxf(mx, __shfl_xor(mx, o, 32));

  // pass 2: denom
  float den = 0.f;
  for (int j = lane; j < d; j += 32) {
    float e = ssrc[g * Nc + edges[j]] + sd;
    e = (e > 0.f) ? e : SLOPE * e;
    den += __expf(e - mx);
  }
#pragma unroll
  for (int o = 16; o > 0; o >>= 1) den += __shfl_xor(den, o, 32);
  const float rden = 1.0f / den;

  // pass 3: weighted aggregation, lanes split the 512 features 32-ways (16 per lane)
  float acc[Fc / 32];
#pragma unroll
  for (int k = 0; k < Fc / 32; ++k) acc[k] = 0.f;
  for (int j = 0; j < d; ++j) {
    int s = edges[j];
    float e = ssrc[g * Nc + s] + sd;
    e = (e > 0.f) ? e : SLOPE * e;
    float wgt = __expf(e - mx) * rden;
    const float* hs = H + (size_t)(g * Nc + s) * Fc;
#pragma unroll
    for (int k = 0; k < Fc / 32; ++k) acc[k] = fmaf(hs[lane + 32 * k], wgt, acc[k]);
  }

  const size_t base = (size_t)w * Fc;
  if (layer == 0) {
#pragma unroll
    for (int k = 0; k < Fc / 32; ++k) {
      int f = lane + 32 * k;
      float x = gelu_exact(acc[k] + bias[f]);
      Ybf[base + f] = (__bf16)x;
    }
  } else {
    int b = g / Tc, t = g % Tc;
    float msk = (t < valid[b]) ? 1.0f : 0.0f;
#pragma unroll
    for (int k = 0; k < Fc / 32; ++k) {
      int f = lane + 32 * k;
      float x = (acc[k] + bias[f]) * msk;
      Yf32[base + f] = x;
      Ybf[base + f]  = (__bf16)x;
    }
  }
}

// ---------------- time mixing: out[b,t,col] = gelu( (Wtime @ Y[b])[t,col] + b_time[t] ) + Y[b,t,col]
// K = T = 32 -> exactly one v_wmma_f32_16x16x32_bf16 per 16x16 tile
__global__ void time_mix_wmma(const __bf16* __restrict__ Wt_bf,   // [T][T] row-major (t,s)
                              const __bf16* __restrict__ Ybf,     // [B][T][N*F] bf16 (masked)
                              const float*  __restrict__ Yf32,    // [B][T][N*F] f32  (masked)
                              const float*  __restrict__ b_time,
                              float* __restrict__ out) {
  constexpr int COLS = Nc * Fc;                       // 65536
  const int lane = threadIdx.x & 31;
  const int half = lane >> 4;
  const int l15  = lane & 15;
  const int wid  = blockIdx.x * (blockDim.x >> 5) + (threadIdx.x >> 5);
  const int wavesPerB = 2 * (COLS / 16);              // 2 t-tiles x 4096 col-tiles
  const int b  = wid / wavesPerB;
  const int r  = wid % wavesPerB;
  const int tt = r / (COLS / 16);
  const int col0 = (r % (COLS / 16)) * 16;

  // A fragment from W_time: A[m][k] = Wtime[tt*16+m][k]
  const __bf16* arow = Wt_bf + (size_t)(tt * 16 + l15) * Tc + half * 8;
  bf16x8 alo = *(const bf16x8*)(arow);
  bf16x8 ahi = *(const bf16x8*)(arow + 16);
  v16bf a;
#pragma unroll
  for (int i = 0; i < 8; ++i) { a[i] = alo[i]; a[8 + i] = ahi[i]; }

  // B fragment: B[k][n] = Ybf[b][s=k][col0+n], n = l15, k in table order
  const __bf16* brow = Ybf + (size_t)b * Tc * COLS + col0 + l15;
  v16bf bb;
#pragma unroll
  for (int i = 0; i < 8; ++i) { bb[i] = brow[(size_t)(half * 8 + i) * COLS]; bb[8 + i] = brow[(size_t)(16 + half * 8 + i) * COLS]; }

  v8f acc = {};
  acc = __builtin_amdgcn_wmma_f32_16x16x32_bf16(false, a, false, bb, (short)0, acc, false, false);

  // epilogue: gelu(z + b_time[t]) + Y
#pragma unroll
  for (int rr = 0; rr < 8; ++rr) {
    int t = tt * 16 + rr + 8 * half;
    size_t idx = ((size_t)b * Tc + t) * COLS + col0 + l15;
    float z = gelu_exact(acc[rr] + b_time[t]);
    out[idx] = z + Yf32[idx];
  }
}

// ---------------- launcher ----------------
extern "C" void kernel_launch(void* const* d_in, const int* in_sizes, int n_in,
                              void* d_out, int out_size, void* d_ws, size_t ws_size,
                              hipStream_t stream) {
  const float* x       = (const float*)d_in[0];
  const int*   ei      = (const int*)  d_in[1];
  const int*   valid   = (const int*)  d_in[2];
  const float* W1      = (const float*)d_in[3];
  const float* a1_src  = (const float*)d_in[4];
  const float* a1_dst  = (const float*)d_in[5];
  const float* b1      = (const float*)d_in[6];
  const float* W2      = (const float*)d_in[7];
  const float* a2_src  = (const float*)d_in[8];
  const float* a2_dst  = (const float*)d_in[9];
  const float* b2      = (const float*)d_in[10];
  const float* W_time  = (const float*)d_in[11];
  const float* b_time  = (const float*)d_in[12];
  float* out = (float*)d_out;

  // workspace layout
  char* ws = (char*)d_ws;
  constexpr size_t NNODE  = (size_t)Gc * Nc;          // 32768
  constexpr size_t NELEM  = NNODE * Fc;               // 16.7M
  constexpr size_t SZ_XBF = NELEM * 2;                // bf16 activations (X -> H1bf -> Ybf)
  constexpr size_t SZ_WT  = (size_t)Fc * Fc * 2;
  constexpr size_t SZ_H   = NELEM * 4;
  constexpr size_t SZ_S   = NNODE * 4;
  constexpr size_t SZ_CSR = (size_t)Gc * EPG * 4;

  size_t o = 0;
  __bf16* Xbf   = (__bf16*)(ws + o); o += SZ_XBF;     // reused as H1bf then Ybf
  __bf16* W1t   = (__bf16*)(ws + o); o += SZ_WT;
  __bf16* W2t   = (__bf16*)(ws + o); o += SZ_WT;
  __bf16* Wtbf  = (__bf16*)(ws + o); o += 4096;
  float*  H     = (float*) (ws + o); o += SZ_H;       // H1 then H2
  float*  Y     = (float*) (ws + o); o += SZ_H;       // masked layer-2 output
  float*  ssrc  = (float*) (ws + o); o += SZ_S;
  float*  sdst  = (float*) (ws + o); o += SZ_S;
  int*    deg   = (int*)   (ws + o); o += SZ_S;
  int*    offv  = (int*)   (ws + o); o += SZ_S;
  int*    cur   = (int*)   (ws + o); o += SZ_S;
  int*    csr   = (int*)   (ws + o); o += SZ_CSR;
  (void)ws_size;

  // 1) conversions
  cvt_f32_to_bf16<<<4096, 256, 0, stream>>>(x, Xbf, NELEM);
  cvt_transpose_bf16<<<1024, 256, 0, stream>>>(W1, W1t);
  cvt_transpose_bf16<<<1024, 256, 0, stream>>>(W2, W2t);
  cvt_f32_to_bf16<<<4, 256, 0, stream>>>(W_time, Wtbf, (size_t)Tc * Tc);

  // 2) CSR (shared by both layers)
  hipMemsetAsync(deg, 0, SZ_S, stream);
  hipMemsetAsync(cur, 0, SZ_S, stream);
  const int nedge_blocks = (Gc * EPG + 255) / 256;
  deg_kernel<<<nedge_blocks, 256, 0, stream>>>(ei, deg);
  scan_kernel<<<Gc, Nc, 0, stream>>>(deg, offv);
  fill_kernel<<<nedge_blocks, 256, 0, stream>>>(ei, offv, cur, csr);

  const int M = (int)NNODE;
  const int gemm_blocks  = (M / 128) * (Fc / 64);     // 2048
  const int node_blocks  = (int)(NNODE / 8);          // 4096 (8 waves/block)

  // 3) layer 1
  gemm_bf16_wmma<<<gemm_blocks, 256, 0, stream>>>(Xbf, W1t, H, M);
  scores_kernel<<<node_blocks, 256, 0, stream>>>(H, a1_src, a1_dst, ssrc, sdst);
  gat_agg_kernel<<<node_blocks, 256, 0, stream>>>(H, ssrc, sdst, offv, deg, csr,
                                                  b1, valid, /*Yf32*/nullptr, Xbf, /*layer*/0);

  // 4) layer 2
  gemm_bf16_wmma<<<gemm_blocks, 256, 0, stream>>>(Xbf, W2t, H, M);
  scores_kernel<<<node_blocks, 256, 0, stream>>>(H, a2_src, a2_dst, ssrc, sdst);
  gat_agg_kernel<<<node_blocks, 256, 0, stream>>>(H, ssrc, sdst, offv, deg, csr,
                                                  b2, valid, Y, Xbf, /*layer*/1);

  // 5) time mixing (WMMA, K=32 in one shot) + gelu + residual -> d_out
  const int time_waves  = Bc * 2 * (Nc * Fc / 16);    // 65536 waves
  time_mix_wmma<<<time_waves / 8, 256, 0, stream>>>(Wtbf, Xbf, Y, b_time, out);
  (void)out_size; (void)n_in; (void)in_sizes;
}
